// DkNN_simple_26620207301314
// MI455X (gfx1250) — matrix operand order repaired and checked by reading, not compiled, not run on hardware
//
#include <hip/hip_runtime.h>
#include <hip/hip_bf16.h>

typedef __attribute__((ext_vector_type(16))) __bf16 v16bf;
typedef __attribute__((ext_vector_type(8)))  __bf16 v8bf;
typedef __attribute__((ext_vector_type(8)))  float  v8f;

#define NQ         1024
#define NTRAIN     50000
#define NTRAIN_PAD 50016     // padded to multiple of 32 (16 dummy rows, norm=1e30)
#define MROWS      51024     // NQ + NTRAIN (real rows)
#define MROWS_PAD  51040     // NQ + NTRAIN_PAD = 16 * 3190
#define K75        75
#define NLAB       8
#define NCH32      1563      // NTRAIN_PAD / 32
#define NSLICE     4
#define CH32_PER_SLICE 391   // ceil(1563/4)

// ---------------------------------------------------------------- helpers

__device__ __forceinline__ v8f wmma_bf16(v16bf a, v16bf b, v8f c) {
    // D = A(16x32 bf16) * B(32x16 bf16) + C(16x16 f32)
    return __builtin_amdgcn_wmma_f32_16x16x32_bf16(
        /*neg_a=*/false, a, /*neg_b=*/false, b,
        /*c_mod=*/(short)0, c, /*reuse_a=*/false, /*reuse_b=*/false);
}

// A-matrix 16x32 bf16 fragment, contiguous k order (B loads use the same):
//   lane<16:  M=lane,    elems 0..15 -> K=ks*32+0..15
//   lane>=16: M=lane-16, elems 0..15 -> K=ks*32+16..31
__device__ __forceinline__ v16bf load_afrag(const __bf16* rowp, int ks, int lane) {
    const int kbase = (lane < 16) ? 0 : 16;
    const __bf16* p = rowp + ks * 32 + kbase;
    v8bf lo = *(const v8bf*)(p);
    v8bf hi = *(const v8bf*)(p + 8);
    v16bf a;
#pragma unroll
    for (int i = 0; i < 8; ++i) { a[i] = lo[i]; a[8 + i] = hi[i]; }
    return a;
}

// ---------------------------------------------------------------- small prep kernels

__global__ void zero_f32_kernel(float* __restrict__ p, int n) {
    int i = blockIdx.x * blockDim.x + threadIdx.x;
    if (i < n) p[i] = 0.0f;
}

// bank0: [MROWS_PAD, 96] bf16; rows 0..1023 = x, 1024..51023 = train_x, rest zero
__global__ void build_bank0_kernel(const float* __restrict__ x,
                                   const float* __restrict__ tx,
                                   __bf16* __restrict__ bank0) {
    int idx = blockIdx.x * blockDim.x + threadIdx.x;
    if (idx >= MROWS_PAD * 96) return;
    int r = idx / 96, c = idx % 96;
    float v = 0.0f;
    if (c < 83 && r < MROWS) v = (r < NQ) ? x[r * 83 + c] : tx[(r - NQ) * 83 + c];
    bank0[idx] = (__bf16)v;
}

// Wt[n][k] = W[k][n]  (W row-major [din,128]) with k >= din zero padded
__global__ void build_wt_kernel(const float* __restrict__ W, int din, int kpad,
                                __bf16* __restrict__ wt) {
    int idx = blockIdx.x * blockDim.x + threadIdx.x;
    if (idx >= 128 * kpad) return;
    int n = idx / kpad, k = idx % kpad;
    wt[idx] = (__bf16)((k < din) ? W[k * 128 + n] : 0.0f);
}

// row norms; dummy pad rows get a huge norm so they never enter a top-75
__global__ void row_norms_kernel(const __bf16* __restrict__ bank, int kpad,
                                 float* __restrict__ out) {
    int r = blockIdx.x * blockDim.x + threadIdx.x;
    if (r >= MROWS_PAD) return;
    const __bf16* p = bank + (size_t)r * kpad;
    float s = 0.0f;
    for (int k = 0; k < kpad; ++k) { float v = (float)p[k]; s += v * v; }
    out[r] = (r < MROWS) ? s : 1.0e30f;
}

// ---------------------------------------------------------------- WMMA MLP layer
// out = relu(in @ W + b), in: [mrows, KSTEPS*32] bf16, wt: [128, KSTEPS*32] bf16
// (transposed W), out: [mrows, 128] bf16. One wave per 16-row M-tile; A register-resident.

template <int KSTEPS>
__global__ __launch_bounds__(128) void mlp_wmma_kernel(
    const __bf16* __restrict__ in_bank,
    const __bf16* __restrict__ wt, const float* __restrict__ bias,
    __bf16* __restrict__ out_bank, int mrows) {
    constexpr int kin = KSTEPS * 32;
    const int lane  = threadIdx.x & 31;
    const int warp  = threadIdx.x >> 5;
    const int mtile = blockIdx.x * 4 + warp;
    if (mtile * 16 >= mrows) return;

    const int rowA = mtile * 16 + (lane & 15);
    v16bf afrag[KSTEPS];
#pragma unroll
    for (int ks = 0; ks < KSTEPS; ++ks)
        afrag[ks] = load_afrag(in_bank + (size_t)rowA * kin, ks, lane);

    const int col   = lane & 15;
    const int rbase = mtile * 16 + ((lane < 16) ? 0 : 8);
    const int koff  = (lane < 16) ? 0 : 16;

#pragma unroll
    for (int nt = 0; nt < 8; ++nt) {
        v8f c = {0.f, 0.f, 0.f, 0.f, 0.f, 0.f, 0.f, 0.f};
        const __bf16* wrow = wt + (size_t)(nt * 16 + col) * kin + koff;
#pragma unroll
        for (int ks = 0; ks < KSTEPS; ++ks) {
            v16bf bfrag = *(const v16bf*)(wrow + ks * 32);
            c = wmma_bf16(afrag[ks], bfrag, c);
        }
        float bv = bias[nt * 16 + col];
#pragma unroll
        for (int v = 0; v < 8; ++v) {
            float x = c[v] + bv;
            x = (x > 0.0f) ? x : 0.0f;
            out_bank[(size_t)(rbase + v) * 128 + nt * 16 + col] = (__bf16)x;
        }
    }
}

// ---------------------------------------------------------------- layer 4 + softmax
// logits = h3 @ W4 + b4 (128->8), softmax, store bank4 [MROWS_PAD, 32] bf16

__global__ void layer4_kernel(const __bf16* __restrict__ h3,
                              const float* __restrict__ W4,
                              const float* __restrict__ b4,
                              __bf16* __restrict__ bank4) {
    __shared__ float sW[128 * NLAB];
    __shared__ float sb[NLAB];
    for (int i = threadIdx.x; i < 128 * NLAB; i += blockDim.x) sW[i] = W4[i];
    if (threadIdx.x < NLAB) sb[threadIdx.x] = b4[threadIdx.x];
    __syncthreads();

    int r = blockIdx.x * blockDim.x + threadIdx.x;
    if (r >= MROWS_PAD) return;

    float acc[NLAB];
#pragma unroll
    for (int c = 0; c < NLAB; ++c) acc[c] = sb[c];
    const __bf16* hp = h3 + (size_t)r * 128;
    for (int k = 0; k < 128; ++k) {
        float h = (float)hp[k];
#pragma unroll
        for (int c = 0; c < NLAB; ++c) acc[c] += h * sW[k * NLAB + c];
    }
    float m = acc[0];
#pragma unroll
    for (int c = 1; c < NLAB; ++c) m = fmaxf(m, acc[c]);
    float s = 0.0f;
#pragma unroll
    for (int c = 0; c < NLAB; ++c) { acc[c] = __expf(acc[c] - m); s += acc[c]; }
    float inv = 1.0f / s;
    __bf16* op = bank4 + (size_t)r * 32;
#pragma unroll
    for (int c = 0; c < NLAB; ++c) op[c] = (__bf16)(acc[c] * inv);
#pragma unroll
    for (int c = NLAB; c < 32; ++c) op[c] = (__bf16)0.0f;
}

// ---------------------------------------------------------------- fused distance GEMM + top-75
// One wave per (16-query tile, N-slice); one launch per layer (KSTEPS templated).
// 32 candidates per iteration (two 16x16 WMMA N-tiles sharing register-resident A).
// Ballot fast path: per-query 75th-best distance lives in LDS; a chunk only enters
// the serial insert phase when some lane's register d2 beats a current worst.
// No block barriers: each wave owns its LDS slice, DS is in-order per wave.

template <int KSTEPS>
__global__ __launch_bounds__(128, 1) void knn_topk_kernel(
    const __bf16* __restrict__ bank,       // [MROWS_PAD, KSTEPS*32]
    const float* __restrict__ nrm,         // [MROWS_PAD]
    const int* __restrict__ train_label,   // [NTRAIN]
    float* __restrict__ topk_d,            // [NQ][NSLICE][K75]
    int* __restrict__ topk_l) {
    constexpr int kpad = KSTEPS * 32;
    __shared__ float s_d2[4][16][33];
    __shared__ float s_hd[4][16][K75];
    __shared__ int   s_hl[4][16][K75];
    __shared__ __align__(32) float s_worst[4][16];

    const int lane  = threadIdx.x & 31;
    const int warp  = threadIdx.x >> 5;
    const int w     = blockIdx.x * 4 + warp;  // 0..255
    const int qtile = w >> 2;
    const int slice = w & 3;
    const int qbase = qtile * 16;
    const int n     = lane & 15;
    const int mh    = (lane < 16) ? 0 : 8;
    const int koff  = (lane < 16) ? 0 : 16;

    // query A-fragments + per-lane query norms, register resident
    v16bf afrag[KSTEPS];
    {
        const __bf16* rowp = bank + (size_t)(qbase + n) * kpad;
#pragma unroll
        for (int ks = 0; ks < KSTEPS; ++ks) afrag[ks] = load_afrag(rowp, ks, lane);
    }
    float qn8[8];
#pragma unroll
    for (int v = 0; v < 8; ++v) qn8[v] = nrm[qbase + mh + v];

    if (lane < 16) s_worst[warp][lane] = 3.4e38f;
    __builtin_amdgcn_wave_barrier();

    int   cnt   = 0;           // per-query state, meaningful for lane<16
    float worst = 3.4e38f;

    const int ch0   = slice * CH32_PER_SLICE;
    const int chEnd = (ch0 + CH32_PER_SLICE < NCH32) ? (ch0 + CH32_PER_SLICE) : NCH32;

    for (int chunk = ch0; chunk < chEnd; ++chunk) {
        const int candBase = chunk * 32;

        // ---- two distance GEMM tiles sharing A: 16 queries x 32 candidates
        v8f c0 = {0.f, 0.f, 0.f, 0.f, 0.f, 0.f, 0.f, 0.f};
        v8f c1 = {0.f, 0.f, 0.f, 0.f, 0.f, 0.f, 0.f, 0.f};
        const __bf16* tb0 = bank + (size_t)(NQ + candBase + n) * kpad + koff;
        const __bf16* tb1 = tb0 + (size_t)16 * kpad;
#pragma unroll
        for (int ks = 0; ks < KSTEPS; ++ks) {
            v16bf bf0 = *(const v16bf*)(tb0 + ks * 32);
            c0 = wmma_bf16(afrag[ks], bf0, c0);
            v16bf bf1 = *(const v16bf*)(tb1 + ks * 32);
            c1 = wmma_bf16(afrag[ks], bf1, c1);
        }

        // prefetch the train bank ~4 chunks ahead (global_prefetch_b8)
        int pf = candBase + 128;
        if (pf + 31 >= NTRAIN_PAD) pf = candBase;
        __builtin_prefetch(bank + (size_t)(NQ + pf + n) * kpad, 0, 1);

        // ---- register d2 + ballot admission test (fast path skips everything)
        const float tn0 = nrm[NQ + candBase + n];
        const float tn1 = nrm[NQ + candBase + 16 + n];
        v8f wv = *(const v8f*)(&s_worst[warp][mh]);   // 32B-aligned LDS vector read
        float d2a[8], d2b[8];
        bool  pass = false;
#pragma unroll
        for (int v = 0; v < 8; ++v) {
            float da = fmaxf(qn8[v] + tn0 - 2.0f * c0[v], 0.0f);
            float db = fmaxf(qn8[v] + tn1 - 2.0f * c1[v], 0.0f);
            d2a[v] = da; d2b[v] = db;
            pass |= (da < wv[v]) | (db < wv[v]);
        }
        if (__ballot(pass) != 0) {                    // wave-uniform slow path
            // stage d2 tiles: (query = mh+v, candidate = t)
#pragma unroll
            for (int v = 0; v < 8; ++v) {
                s_d2[warp][mh + v][n]      = d2a[v];
                s_d2[warp][mh + v][16 + n] = d2b[v];
            }
            __builtin_amdgcn_wave_barrier();

            if (lane < 16) {                          // lane == query row
                float* hd = s_hd[warp][lane];
                int*   hl = s_hl[warp][lane];
                for (int t = 0; t < 32; ++t) {
                    float d2 = s_d2[warp][lane][t];
                    if (cnt < K75 || d2 < worst) {
                        int cand = candBase + t;
                        int lab = (cand < NTRAIN) ? train_label[cand] : 0;
                        int j = (cnt < K75) ? cnt : (K75 - 1);
                        while (j > 0 && hd[j - 1] > d2) {
                            hd[j] = hd[j - 1]; hl[j] = hl[j - 1]; --j;
                        }
                        hd[j] = d2; hl[j] = lab;
                        if (cnt < K75) ++cnt;
                        if (cnt == K75) worst = hd[K75 - 1];
                    }
                }
                s_worst[warp][lane] = worst;
            }
            __builtin_amdgcn_wave_barrier();
        }
    }

    if (lane < 16) {
        size_t obase = ((size_t)(qbase + lane) * NSLICE + slice) * (size_t)K75;
        for (int t = 0; t < K75; ++t) {
            topk_d[obase + t] = s_hd[warp][lane][t];
            topk_l[obase + t] = s_hl[warp][lane][t];
        }
    }
}

// ---------------------------------------------------------------- merge slices -> nonconformity

__global__ void knn_merge_kernel(const float* __restrict__ topk_d,
                                 const int* __restrict__ topk_l,
                                 float* __restrict__ totalNC) {
    int idx = blockIdx.x * blockDim.x + threadIdx.x;   // (layer, query)
    if (idx >= 5 * NQ) return;
    int q = idx % NQ;
    size_t base = (size_t)idx * NSLICE * K75;

    int   p[NSLICE] = {0, 0, 0, 0};
    float total = 0.0f;
    float wc[NLAB] = {0.f, 0.f, 0.f, 0.f, 0.f, 0.f, 0.f, 0.f};

    for (int t = 0; t < K75; ++t) {                    // 4-way merge of sorted lists
        float bd = 3.4e38f;
        int   bs = 0;
#pragma unroll
        for (int s = 0; s < NSLICE; ++s) {
            if (p[s] < K75) {
                float d = topk_d[base + s * K75 + p[s]];
                if (d < bd) { bd = d; bs = s; }
            }
        }
        int lab = topk_l[base + bs * K75 + p[bs]] & 7;
        ++p[bs];
        float wv = (bd > 0.0f) ? rsqrtf(bd) : 0.0f;    // 1/dist weighting
        total += wv;
        wc[lab] += wv;
    }
#pragma unroll
    for (int c = 0; c < NLAB; ++c)
        atomicAdd(&totalNC[q * NLAB + c], total - wc[c]);
}

// ---------------------------------------------------------------- empirical p-values

__global__ void pvalue_kernel(const float* __restrict__ totalNC,
                              const float* __restrict__ cali, int C,
                              float* __restrict__ out) {
    __shared__ int sc[NLAB];
    const int q = blockIdx.x;
    if (threadIdx.x < NLAB) sc[threadIdx.x] = 0;
    __syncthreads();

    float th[NLAB];
#pragma unroll
    for (int c = 0; c < NLAB; ++c) th[c] = totalNC[q * NLAB + c];

    int loc[NLAB] = {0, 0, 0, 0, 0, 0, 0, 0};
    for (int i = threadIdx.x; i < C; i += blockDim.x) {
        float cv = cali[i];
#pragma unroll
        for (int c = 0; c < NLAB; ++c) loc[c] += (cv >= th[c]) ? 1 : 0;
    }
#pragma unroll
    for (int c = 0; c < NLAB; ++c) atomicAdd(&sc[c], loc[c]);
    __syncthreads();
    if (threadIdx.x < NLAB)
        out[q * NLAB + threadIdx.x] = (float)sc[threadIdx.x] / (float)C;
}

// ---------------------------------------------------------------- launch

extern "C" void kernel_launch(void* const* d_in, const int* in_sizes, int n_in,
                              void* d_out, int out_size, void* d_ws, size_t ws_size,
                              hipStream_t stream) {
    const float* x    = (const float*)d_in[0];
    const float* tx   = (const float*)d_in[1];
    const int*   tlab = (const int*)d_in[2];
    const float* cali = (const float*)d_in[3];
    const float* W1   = (const float*)d_in[4];
    const float* b1   = (const float*)d_in[5];
    const float* W2   = (const float*)d_in[6];
    const float* b2   = (const float*)d_in[7];
    const float* W3   = (const float*)d_in[8];
    const float* b3   = (const float*)d_in[9];
    const float* W4   = (const float*)d_in[10];
    const float* b4   = (const float*)d_in[11];
    const int C = in_sizes[3];

    size_t off = 0;
    auto take = [&](size_t bytes) -> char* {
        char* p = (char*)d_ws + off;
        off += (bytes + 255) & ~(size_t)255;
        return p;
    };
    __bf16* bank0 = (__bf16*)take((size_t)MROWS_PAD * 96 * 2);
    __bf16* bank1 = (__bf16*)take((size_t)MROWS_PAD * 128 * 2);
    __bf16* bank2 = (__bf16*)take((size_t)MROWS_PAD * 128 * 2);
    __bf16* bank3 = (__bf16*)take((size_t)MROWS_PAD * 128 * 2);
    __bf16* bank4 = (__bf16*)take((size_t)MROWS_PAD * 32 * 2);
    __bf16* wt1   = (__bf16*)take((size_t)128 * 96 * 2);
    __bf16* wt2   = (__bf16*)take((size_t)128 * 128 * 2);
    __bf16* wt3   = (__bf16*)take((size_t)128 * 128 * 2);
    float*  norms = (float*)take((size_t)5 * MROWS_PAD * 4);
    float*  totalNC = (float*)take((size_t)NQ * NLAB * 4);
    float*  topk_d  = (float*)take((size_t)5 * NQ * NSLICE * K75 * 4);
    int*    topk_l  = (int*)take((size_t)5 * NQ * NSLICE * K75 * 4);
    (void)ws_size; (void)n_in; (void)out_size;

    // prep
    zero_f32_kernel<<<(NQ * NLAB + 255) / 256, 256, 0, stream>>>(totalNC, NQ * NLAB);
    build_bank0_kernel<<<(MROWS_PAD * 96 + 255) / 256, 256, 0, stream>>>(x, tx, bank0);
    build_wt_kernel<<<(128 * 96 + 255) / 256, 256, 0, stream>>>(W1, 83, 96, wt1);
    build_wt_kernel<<<(128 * 128 + 255) / 256, 256, 0, stream>>>(W2, 128, 128, wt2);
    build_wt_kernel<<<(128 * 128 + 255) / 256, 256, 0, stream>>>(W3, 128, 128, wt3);

    // MLP (queries + train + pad rows jointly: 51040 rows = 3190 M-tiles)
    const int mlp_blocks = (MROWS_PAD / 16 + 3) / 4;  // 798
    mlp_wmma_kernel<3><<<mlp_blocks, 128, 0, stream>>>(bank0, wt1, b1, bank1, MROWS_PAD);
    mlp_wmma_kernel<4><<<mlp_blocks, 128, 0, stream>>>(bank1, wt2, b2, bank2, MROWS_PAD);
    mlp_wmma_kernel<4><<<mlp_blocks, 128, 0, stream>>>(bank2, wt3, b3, bank3, MROWS_PAD);
    layer4_kernel<<<(MROWS_PAD + 255) / 256, 256, 0, stream>>>(bank3, W4, b4, bank4);

    // row norms for all 5 banks (pad rows forced to 1e30)
    const int ngrid = (MROWS_PAD + 255) / 256;
    row_norms_kernel<<<ngrid, 256, 0, stream>>>(bank0, 96,  norms + 0 * MROWS_PAD);
    row_norms_kernel<<<ngrid, 256, 0, stream>>>(bank1, 128, norms + 1 * MROWS_PAD);
    row_norms_kernel<<<ngrid, 256, 0, stream>>>(bank2, 128, norms + 2 * MROWS_PAD);
    row_norms_kernel<<<ngrid, 256, 0, stream>>>(bank3, 128, norms + 3 * MROWS_PAD);
    row_norms_kernel<<<ngrid, 256, 0, stream>>>(bank4, 32,  norms + 4 * MROWS_PAD);

    // fused distance GEMM + top-k: per layer, 64 q-tiles x 4 slices = 256 waves = 64 blocks
    const size_t lstride = (size_t)NQ * NSLICE * K75;
    knn_topk_kernel<3><<<64, 128, 0, stream>>>(bank0, norms + 0 * MROWS_PAD, tlab,
                                               topk_d + 0 * lstride, topk_l + 0 * lstride);
    knn_topk_kernel<4><<<64, 128, 0, stream>>>(bank1, norms + 1 * MROWS_PAD, tlab,
                                               topk_d + 1 * lstride, topk_l + 1 * lstride);
    knn_topk_kernel<4><<<64, 128, 0, stream>>>(bank2, norms + 2 * MROWS_PAD, tlab,
                                               topk_d + 2 * lstride, topk_l + 2 * lstride);
    knn_topk_kernel<4><<<64, 128, 0, stream>>>(bank3, norms + 3 * MROWS_PAD, tlab,
                                               topk_d + 3 * lstride, topk_l + 3 * lstride);
    knn_topk_kernel<1><<<64, 128, 0, stream>>>(bank4, norms + 4 * MROWS_PAD, tlab,
                                               topk_d + 4 * lstride, topk_l + 4 * lstride);

    knn_merge_kernel<<<(5 * NQ + 255) / 256, 256, 0, stream>>>(topk_d, topk_l, totalNC);
    pvalue_kernel<<<NQ, 256, 0, stream>>>(totalNC, cali, C, (float*)d_out);
}
